// TextPromptNetwork_91122026152467
// MI455X (gfx1250) — matrix operand order repaired and checked by reading, not compiled, not run on hardware
//
#include <hip/hip_runtime.h>
#include <math.h>

typedef __attribute__((ext_vector_type(16))) _Float16 v16h;
typedef __attribute__((ext_vector_type(8)))  float    v8f;
typedef __attribute__((ext_vector_type(4)))  unsigned int u32x4;
typedef __attribute__((ext_vector_type(8)))  int          i32x8;
typedef __attribute__((ext_vector_type(4)))  int          i32x4;

#define SEQ     2048
#define DMODEL  768
#define NHEADS  8
#define QRANK   64
#define KVRANK  64
#define ROPED   64
#define NOPED   32
#define VDIM    96
#define QHD     96
#define EPSV    1e-6f

// ---------------------------------------------------------------- helpers ---

__device__ __forceinline__ v8f wmma_f16(v16h a, v16h b, v8f c) {
  return __builtin_amdgcn_wmma_f32_16x16x32_f16(false, a, false, b, (short)0, c,
                                                false, false);
}

__device__ __forceinline__ v8f vzero8() {
  v8f z;
#pragma unroll
  for (int i = 0; i < 8; ++i) z[i] = 0.0f;
  return z;
}

// A-fragment (16x32, MxK): lane = 16*half + m; holds K = 16*half + i
__device__ __forceinline__ v16h ld_a_f16(const _Float16* base, int ld, int k0) {
  const int l = threadIdx.x & 31, h = l >> 4, m = l & 15;
  const _Float16* p = base + (size_t)m * ld + k0 + h * 16;
  v16h a;
#pragma unroll
  for (int i = 0; i < 16; ++i) a[i] = p[i];
  return a;
}
// B-fragment (32x16, KxN): lane = 16*half + n; holds K = 16*half + i.
// Source is "N rows x K cols" row-major: b[i] = base[n*ld + k0 + 16*half + i]
__device__ __forceinline__ v16h ld_b_f16(const _Float16* base, size_t ld, int k0) {
  const int l = threadIdx.x & 31, h = l >> 4, n = l & 15;
  const _Float16* p = base + (size_t)n * ld + k0 + h * 16;
  v16h b;
#pragma unroll
  for (int i = 0; i < 16; ++i) b[i] = p[i];
  return b;
}

__device__ __forceinline__ float hmax16(float v) {
#pragma unroll
  for (int o = 1; o < 16; o <<= 1) v = fmaxf(v, __shfl_xor(v, o, 32));
  return v;
}
__device__ __forceinline__ float hsum16(float v) {
#pragma unroll
  for (int o = 1; o < 16; o <<= 1) v += __shfl_xor(v, o, 32);
  return v;
}
__device__ __forceinline__ float hsum32(float v) {
#pragma unroll
  for (int o = 1; o < 32; o <<= 1) v += __shfl_xor(v, o, 32);
  return v;
}

// wave-private LDS producer->consumer fence (LDS in-order per wave)
__device__ __forceinline__ void wave_lds_fence() {
  __builtin_amdgcn_wave_barrier();
  asm volatile("s_wait_dscnt 0" ::: "memory");
  __builtin_amdgcn_wave_barrier();
}

// --- Tensor Data Mover: build D# groups per CDNA5 ISA 8.3/8.4 and issue -----
__device__ __forceinline__ unsigned lds_byte_off(const void* p) {
  return (unsigned)(unsigned long long)p;  // low 32 bits of LDS flat address
}

__device__ __forceinline__ void tdm_load_to_lds(unsigned lds_off,
                                                unsigned long long ga,
                                                unsigned tile0, unsigned tile1,
                                                unsigned long long td0,
                                                unsigned long long td1,
                                                unsigned long long stride0) {
  u32x4 g0;
  g0[0] = 1u;                                   // count=1, user mode, no gather
  g0[1] = lds_off;                              // lds_addr [63:32]
  g0[2] = (unsigned)ga;                         // global_addr [95:64]
  g0[3] = (unsigned)((ga >> 32) & 0x01ffffffu)  // global_addr [120:96]
        | (2u << 30);                           // type=2 (image) [127:126]
  i32x8 g1;
  g1[0] = (int)(1u << 16);                      // wg_mask=0, data_size=1 (2B)
  g1[1] = (int)((td0 & 0xffffu) << 16);                       // tensor_dim0 lo
  g1[2] = (int)(((td0 >> 16) & 0xffffu) | ((td1 & 0xffffu) << 16));
  g1[3] = (int)(((td1 >> 16) & 0xffffu) | ((tile0 & 0xffffu) << 16));
  g1[4] = (int)(tile1 & 0xffffu);               // tile_dim1, tile_dim2=0
  g1[5] = (int)(stride0 & 0xffffffffu);         // tensor_dim0_stride lo
  g1[6] = (int)((stride0 >> 32) & 0xffffu);     // stride hi, dim1_stride=0
  g1[7] = 0;
  i32x4 gz = {0, 0, 0, 0};
#if __clang_major__ >= 23
  i32x8 gz8 = {0, 0, 0, 0, 0, 0, 0, 0};
  __builtin_amdgcn_tensor_load_to_lds(g0, g1, gz, gz, gz8, 0);
#else
  __builtin_amdgcn_tensor_load_to_lds(g0, g1, gz, gz, 0);
#endif
}

// RoPE using precomputed tables (reference permute:
// out[j]=x_even*c-x_odd*s ; out[32+j]=x_odd*c+x_even*s)
__device__ __forceinline__ void rope_store(const float* row, _Float16* dst,
                                           const float* csr, const float* snr,
                                           int part) {
#pragma unroll
  for (int j = 0; j < 32; ++j) {
    float cs = csr[j], sn = snr[j];
    float xe = row[2 * j], xo = row[2 * j + 1];
    float o = part ? (xo * cs + xe * sn) : (xe * cs - xo * sn);
    dst[part * 32 + j] = (_Float16)o;
  }
}

// ---------------------------------------------------------------- kernels ---

// rope tables: cs/sn[pos*32+j] = cos/sin(pos * 10000^(-j/32))
__global__ void __launch_bounds__(256)
k_ropetab(float* __restrict__ cs, float* __restrict__ sn) {
  int i = blockIdx.x * 256 + threadIdx.x;  // SEQ*32
  int pos = i >> 5, j = i & 31;
  const float L2B = 13.287712379549449f;  // log2(10000)
  float inv = exp2f(-((float)j * (1.0f / 32.0f)) * L2B);
  float s, c;
  sincosf((float)pos * inv, &s, &c);
  cs[i] = c;
  sn[i] = s;
}

// f32 -> f16 bulk convert (4 per thread)
__global__ void __launch_bounds__(256)
k_cvt(const float* __restrict__ x, _Float16* __restrict__ y, int n) {
  int i = (blockIdx.x * 256 + threadIdx.x) * 4;
#pragma unroll
  for (int k = 0; k < 4; ++k)
    if (i + k < n) y[i + k] = (_Float16)x[i + k];
}

// q_absorb transpose: absT[h][r][d] = wkv_b[h*128+d][r]  (h<8, d<32, r<64)
__global__ void __launch_bounds__(256)
k_absT(const float* __restrict__ wkv_b, _Float16* __restrict__ absT) {
  int i = blockIdx.x * 256 + threadIdx.x;  // 16384 total
  int d = i & 31, r = (i >> 5) & 63, h = i >> 11;
  absT[i] = (_Float16)wkv_b[(h * 128 + d) * 64 + r];
}

// ckv = src @ wkv_a^T ; rms_norm -> c_lat (S,64)+c_latT (64,NTOK) ; rope -> k_pe
__global__ void __launch_bounds__(256)
k_ckv(const _Float16* __restrict__ src_h, const _Float16* __restrict__ wkv_a_h,
      const float* __restrict__ kv_ln, const float* __restrict__ cs,
      const float* __restrict__ sn, _Float16* __restrict__ c_lat,
      _Float16* __restrict__ c_latT, _Float16* __restrict__ k_pe, int ntok) {
  __shared__ float lds[8][16 * 64];
  const int wave = threadIdx.x >> 5, lane = threadIdx.x & 31;
  const int hh = lane >> 4, nn = lane & 15;
  const int row0 = (blockIdx.x * 8 + wave) * 16;
  const _Float16* A = src_h + (size_t)row0 * DMODEL;
  v8f acc[8];
#pragma unroll
  for (int f = 0; f < 8; ++f) acc[f] = vzero8();
  for (int k0 = 0; k0 < DMODEL; k0 += 32) {
    v16h a = ld_a_f16(A, DMODEL, k0);
#pragma unroll
    for (int f = 0; f < 8; ++f)
      acc[f] = wmma_f16(a, ld_b_f16(wkv_a_h + (size_t)(f * 16) * DMODEL, DMODEL, k0), acc[f]);
  }
  float rsj[8];
#pragma unroll
  for (int j = 0; j < 8; ++j) {
    float s = 0.0f;
#pragma unroll
    for (int f = 0; f < 4; ++f) { float v = acc[f][j]; s += v * v; }
    rsj[j] = rsqrtf(hsum16(s) * (1.0f / KVRANK) + EPSV);
  }
#pragma unroll
  for (int j = 0; j < 8; ++j) {
    int t = row0 + hh * 8 + j;
#pragma unroll
    for (int f = 0; f < 4; ++f) {
      int r = f * 16 + nn;
      float v = acc[f][j] * rsj[j] * kv_ln[r];
      c_lat[(size_t)t * KVRANK + r] = (_Float16)v;
      c_latT[(size_t)r * ntok + t] = (_Float16)v;
    }
#pragma unroll
    for (int f = 4; f < 8; ++f)
      lds[wave][(hh * 8 + j) * 64 + (f - 4) * 16 + nn] = acc[f][j];
  }
  wave_lds_fence();
  {
    int m = lane & 15, part = lane >> 4;
    int t = row0 + m, pos = t & (SEQ - 1);
    rope_store(&lds[wave][m * 64], k_pe + (size_t)t * ROPED,
               cs + pos * 32, sn + pos * 32, part);
  }
}

// q_a = rms_norm(src @ wq_a^T) -> q_a_n (f16)
__global__ void __launch_bounds__(256)
k_qa(const _Float16* __restrict__ src_h, const _Float16* __restrict__ wq_a_h,
     const float* __restrict__ q_ln, _Float16* __restrict__ q_a_n, int ntok) {
  const int wave = threadIdx.x >> 5, lane = threadIdx.x & 31;
  const int hh = lane >> 4, nn = lane & 15;
  const int row0 = (blockIdx.x * 8 + wave) * 16;
  const _Float16* A = src_h + (size_t)row0 * DMODEL;
  v8f acc[4];
#pragma unroll
  for (int f = 0; f < 4; ++f) acc[f] = vzero8();
  for (int k0 = 0; k0 < DMODEL; k0 += 32) {
    v16h a = ld_a_f16(A, DMODEL, k0);
#pragma unroll
    for (int f = 0; f < 4; ++f)
      acc[f] = wmma_f16(a, ld_b_f16(wq_a_h + (size_t)(f * 16) * DMODEL, DMODEL, k0), acc[f]);
  }
#pragma unroll
  for (int j = 0; j < 8; ++j) {
    float s = 0.0f;
#pragma unroll
    for (int f = 0; f < 4; ++f) { float v = acc[f][j]; s += v * v; }
    float rs = rsqrtf(hsum16(s) * (1.0f / QRANK) + EPSV);
    int t = row0 + hh * 8 + j;
#pragma unroll
    for (int f = 0; f < 4; ++f) {
      int r = f * 16 + nn;
      q_a_n[(size_t)t * QRANK + r] = (_Float16)(acc[f][j] * rs * q_ln[r]);
    }
  }
}

// per-head: q = q_a_n @ wq_b^T ; rope(q_pe) ; q_lat = q_nope @ q_absorb
__global__ void __launch_bounds__(256)
k_qb(const _Float16* __restrict__ q_a_n, const _Float16* __restrict__ wq_b_h,
     const _Float16* __restrict__ absT, const float* __restrict__ cs,
     const float* __restrict__ sn, _Float16* __restrict__ q_lat,
     _Float16* __restrict__ q_pe, int ntok) {
  __shared__ _Float16 qn[8][16 * 32];
  __shared__ float pe[8][16 * 64];
  const int wave = threadIdx.x >> 5, lane = threadIdx.x & 31;
  const int hh = lane >> 4, nn = lane & 15;
  const int idx = blockIdx.x * 8 + wave;
  const int h = idx & 7, row0 = (idx >> 3) * 16;
  v8f acc[6];
#pragma unroll
  for (int f = 0; f < 6; ++f) acc[f] = vzero8();
#pragma unroll
  for (int ks = 0; ks < 2; ++ks) {
    v16h a = ld_a_f16(q_a_n + (size_t)row0 * QRANK, QRANK, ks * 32);
#pragma unroll
    for (int f = 0; f < 6; ++f)
      acc[f] = wmma_f16(a, ld_b_f16(wq_b_h + (size_t)(h * QHD + f * 16) * QRANK, QRANK, ks * 32), acc[f]);
  }
#pragma unroll
  for (int j = 0; j < 8; ++j) {
    int rr = hh * 8 + j;
    qn[wave][rr * 32 + nn]      = (_Float16)acc[0][j];
    qn[wave][rr * 32 + 16 + nn] = (_Float16)acc[1][j];
#pragma unroll
    for (int f = 2; f < 6; ++f) pe[wave][rr * 64 + (f - 2) * 16 + nn] = acc[f][j];
  }
  wave_lds_fence();
  {
    int m = lane & 15, part = lane >> 4;
    int t = row0 + m, bb = t >> 11, ss = t & (SEQ - 1);
    rope_store(&pe[wave][m * 64],
               q_pe + ((size_t)(bb * NHEADS + h) * SEQ + ss) * ROPED,
               cs + ss * 32, sn + ss * 32, part);
  }
  v16h a2 = ld_a_f16(qn[wave], 32, 0);
  v8f alat[4];
#pragma unroll
  for (int f = 0; f < 4; ++f) alat[f] = vzero8();
#pragma unroll
  for (int f = 0; f < 4; ++f)
    alat[f] = wmma_f16(a2, ld_b_f16(absT + (size_t)h * 64 * 32 + (size_t)f * 16 * 32, 32, 0), alat[f]);
#pragma unroll
  for (int j = 0; j < 8; ++j) {
    int t = row0 + hh * 8 + j, bb = t >> 11, ss = t & (SEQ - 1);
#pragma unroll
    for (int f = 0; f < 4; ++f)
      q_lat[((size_t)(bb * NHEADS + h) * SEQ + ss) * KVRANK + f * 16 + nn] = (_Float16)alat[f][j];
  }
}

// flash attention: one workgroup = one (b,h) x 128 queries; key tiles staged
// into LDS by the Tensor Data Mover, double-buffered.
__global__ void __launch_bounds__(256)
k_attn(const _Float16* __restrict__ q_lat, const _Float16* __restrict__ q_pe,
       const _Float16* __restrict__ c_lat, const _Float16* __restrict__ c_latT,
       const _Float16* __restrict__ k_pe, const _Float16* __restrict__ wkv_b_h,
       _Float16* __restrict__ attnctx, int ntok) {
  __shared__ _Float16 s_cl[2][32 * 64];  // c_lat key tile (t-local rows, 64 r)
  __shared__ _Float16 s_kp[2][32 * 64];  // k_pe key tile
  __shared__ _Float16 s_cT[2][64 * 32];  // c_latT tile (64 r rows, 32 t cols)
  __shared__ _Float16 pl[8][16 * 32];
  __shared__ _Float16 ctl[8][16 * 64];
  const int wave = threadIdx.x >> 5, lane = threadIdx.x & 31;
  const int hh = lane >> 4, nn = lane & 15;
  const int bh = blockIdx.x >> 4;  // (b*8+h)
  const int sgrp = blockIdx.x & 15;
  const int b = bh >> 3, h = bh & 7;
  const int s0 = (sgrp * 8 + wave) * 16;
  const float scale = 0.10206207261596575f;  // 96^-0.5
  const _Float16* qlb = q_lat + ((size_t)bh * SEQ + s0) * KVRANK;
  const _Float16* qpb = q_pe + ((size_t)bh * SEQ + s0) * ROPED;
  const _Float16* clb0 = c_lat + (size_t)b * SEQ * KVRANK;
  const _Float16* kpb0 = k_pe + (size_t)b * SEQ * ROPED;
  const _Float16* cTb0 = c_latT + (size_t)b * SEQ;
  v16h aql[2], aqp[2];
#pragma unroll
  for (int ks = 0; ks < 2; ++ks) {
    aql[ks] = ld_a_f16(qlb, KVRANK, ks * 32);
    aqp[ks] = ld_a_f16(qpb, ROPED, ks * 32);
  }
  v8f cacc[4];
#pragma unroll
  for (int f = 0; f < 4; ++f) cacc[f] = vzero8();
  float mrun[8], lrun[8];
#pragma unroll
  for (int j = 0; j < 8; ++j) { mrun[j] = -3.0e38f; lrun[j] = 0.0f; }

  auto issue = [&](int buf, int t0) {
    // two contiguous 32x64 f16 tiles (4 KB each) + one strided 64x32 tile
    tdm_load_to_lds(lds_byte_off(&s_cl[buf][0]),
                    (unsigned long long)(const void*)(clb0 + (size_t)t0 * KVRANK),
                    2048, 1, 2048, 1, 2048);
    tdm_load_to_lds(lds_byte_off(&s_kp[buf][0]),
                    (unsigned long long)(const void*)(kpb0 + (size_t)t0 * ROPED),
                    2048, 1, 2048, 1, 2048);
    tdm_load_to_lds(lds_byte_off(&s_cT[buf][0]),
                    (unsigned long long)(const void*)(cTb0 + t0),
                    32, 64, 32, 64, (unsigned long long)ntok);
  };
  if (wave == 0) {
    issue(0, 0);
    __builtin_amdgcn_s_wait_tensorcnt(0);
  }
  const int NT = SEQ / 32;
  for (int it = 0; it < NT; ++it) {
    const int cur = it & 1;
    __syncthreads();  // cur buffer valid; prev buffer free for refill
    if (wave == 0 && it + 1 < NT) issue(cur ^ 1, (it + 1) * 32);

    v8f sc0 = vzero8(), sc1 = vzero8();
#pragma unroll
    for (int ks = 0; ks < 2; ++ks) {
      sc0 = wmma_f16(aql[ks], ld_b_f16(s_cl[cur], 64, ks * 32), sc0);
      sc1 = wmma_f16(aql[ks], ld_b_f16(s_cl[cur] + 16 * 64, 64, ks * 32), sc1);
      sc0 = wmma_f16(aqp[ks], ld_b_f16(s_kp[cur], 64, ks * 32), sc0);
      sc1 = wmma_f16(aqp[ks], ld_b_f16(s_kp[cur] + 16 * 64, 64, ks * 32), sc1);
    }
#pragma unroll
    for (int j = 0; j < 8; ++j) {
      float v0 = sc0[j] * scale, v1 = sc1[j] * scale;
      float nm = fmaxf(mrun[j], hmax16(fmaxf(v0, v1)));
      float al = __expf(mrun[j] - nm);
      float p0 = __expf(v0 - nm), p1 = __expf(v1 - nm);
      lrun[j] = lrun[j] * al + hsum16(p0 + p1);
      mrun[j] = nm;
#pragma unroll
      for (int f = 0; f < 4; ++f) cacc[f][j] *= al;
      int rr = hh * 8 + j;
      pl[wave][rr * 32 + nn]      = (_Float16)p0;
      pl[wave][rr * 32 + 16 + nn] = (_Float16)p1;
    }
    wave_lds_fence();
    v16h pa = ld_a_f16(pl[wave], 32, 0);
#pragma unroll
    for (int f = 0; f < 4; ++f)
      cacc[f] = wmma_f16(pa, ld_b_f16(s_cT[cur] + f * 16 * 32, 32, 0), cacc[f]);
    wave_lds_fence();
    if (wave == 0) __builtin_amdgcn_s_wait_tensorcnt(0);
  }
  // normalize, bounce ctx to LDS, out-absorb GEMM
#pragma unroll
  for (int j = 0; j < 8; ++j) {
    float inv = 1.0f / lrun[j];
    int rr = hh * 8 + j;
#pragma unroll
    for (int f = 0; f < 4; ++f)
      ctl[wave][rr * 64 + f * 16 + nn] = (_Float16)(cacc[f][j] * inv);
  }
  wave_lds_fence();
  v8f o[6];
#pragma unroll
  for (int f = 0; f < 6; ++f) o[f] = vzero8();
#pragma unroll
  for (int ks = 0; ks < 2; ++ks) {
    v16h a2 = ld_a_f16(ctl[wave], 64, ks * 32);
#pragma unroll
    for (int f = 0; f < 6; ++f)
      o[f] = wmma_f16(a2,
          ld_b_f16(wkv_b_h + (size_t)(h * 128 + NOPED + f * 16) * KVRANK, KVRANK, ks * 32), o[f]);
  }
#pragma unroll
  for (int j = 0; j < 8; ++j) {
    int t = b * SEQ + s0 + hh * 8 + j;
#pragma unroll
    for (int f = 0; f < 6; ++f)
      attnctx[(size_t)t * DMODEL + h * VDIM + f * 16 + nn] = (_Float16)o[f][j];
  }
}

// y = src + attnctx @ wo^T
__global__ void __launch_bounds__(256)
k_wo(const _Float16* __restrict__ attnctx, const _Float16* __restrict__ wo_h,
     const float* __restrict__ src, float* __restrict__ y, int ntok) {
  const int wave = threadIdx.x >> 5, lane = threadIdx.x & 31;
  const int hh = lane >> 4, nn = lane & 15;
  const int idx = blockIdx.x * 8 + wave;
  const int row0 = (idx / 12) * 16, n0 = (idx % 12) * 64;
  const _Float16* A = attnctx + (size_t)row0 * DMODEL;
  v8f acc[4];
#pragma unroll
  for (int f = 0; f < 4; ++f) acc[f] = vzero8();
  for (int k0 = 0; k0 < DMODEL; k0 += 32) {
    v16h a = ld_a_f16(A, DMODEL, k0);
#pragma unroll
    for (int f = 0; f < 4; ++f)
      acc[f] = wmma_f16(a, ld_b_f16(wo_h + (size_t)(n0 + f * 16) * DMODEL, DMODEL, k0), acc[f]);
  }
#pragma unroll
  for (int j = 0; j < 8; ++j) {
    int t = row0 + hh * 8 + j;
#pragma unroll
    for (int f = 0; f < 4; ++f) {
      size_t off = (size_t)t * DMODEL + n0 + f * 16 + nn;
      y[off] = src[off] + acc[f][j];
    }
  }
}

// out = rms_norm(y) * norm_w
__global__ void __launch_bounds__(256)
k_outnorm(const float* __restrict__ y, const float* __restrict__ nw,
          float* __restrict__ out, int ntok) {
  const int wave = threadIdx.x >> 5, lane = threadIdx.x & 31;
  const int row = blockIdx.x * 8 + wave;
  const float* yr = y + (size_t)row * DMODEL;
  float ss = 0.0f;
  for (int i = lane; i < DMODEL; i += 32) { float v = yr[i]; ss += v * v; }
  float rs = rsqrtf(hsum32(ss) * (1.0f / DMODEL) + EPSV);
  for (int i = lane; i < DMODEL; i += 32)
    out[(size_t)row * DMODEL + i] = yr[i] * rs * nw[i];
}

// ------------------------------------------------------------------ launch ---

extern "C" void kernel_launch(void* const* d_in, const int* in_sizes, int n_in,
                              void* d_out, int out_size, void* d_ws, size_t ws_size,
                              hipStream_t stream) {
  const float* src      = (const float*)d_in[0];
  const float* wq_a     = (const float*)d_in[1];
  const float* q_a_ln_w = (const float*)d_in[2];
  const float* wq_b     = (const float*)d_in[3];
  const float* wkv_a    = (const float*)d_in[4];
  const float* kv_ln_w  = (const float*)d_in[5];
  const float* wkv_b    = (const float*)d_in[6];
  const float* wo       = (const float*)d_in[7];
  const float* norm_w   = (const float*)d_in[8];
  float* out = (float*)d_out;

  const int ntok = in_sizes[0] / DMODEL;  // B*S = 8192

  char* base = (char*)d_ws;
  size_t off = 0;
  auto alloc = [&](size_t bytes) -> char* {
    char* p = base + off;
    off = (off + bytes + 255) & ~(size_t)255;
    return p;
  };
  _Float16* src_h   = (_Float16*)alloc((size_t)ntok * DMODEL * 2);
  _Float16* wq_a_h  = (_Float16*)alloc((size_t)QRANK * DMODEL * 2);
  _Float16* wq_b_h  = (_Float16*)alloc((size_t)NHEADS * QHD * QRANK * 2);
  _Float16* wkv_a_h = (_Float16*)alloc((size_t)(KVRANK + ROPED) * DMODEL * 2);
  _Float16* wkv_b_h = (_Float16*)alloc((size_t)NHEADS * 128 * KVRANK * 2);
  _Float16* wo_h    = (_Float16*)alloc((size_t)DMODEL * DMODEL * 2);
  _Float16* absT    = (_Float16*)alloc((size_t)NHEADS * 64 * 32 * 2);
  float*    cs_tab  = (float*)alloc((size_t)SEQ * 32 * 4);
  float*    sn_tab  = (float*)alloc((size_t)SEQ * 32 * 4);
  _Float16* c_lat   = (_Float16*)alloc((size_t)ntok * KVRANK * 2);
  _Float16* c_latT  = (_Float16*)alloc((size_t)KVRANK * ntok * 2);
  _Float16* k_pe    = (_Float16*)alloc((size_t)ntok * ROPED * 2);
  _Float16* q_a_n   = (_Float16*)alloc((size_t)ntok * QRANK * 2);
  _Float16* q_lat   = (_Float16*)alloc((size_t)ntok * NHEADS * KVRANK * 2);
  _Float16* q_pe    = (_Float16*)alloc((size_t)ntok * NHEADS * ROPED * 2);
  _Float16* attnctx = (_Float16*)alloc((size_t)ntok * DMODEL * 2);
  float*    yres    = (float*)alloc((size_t)ntok * DMODEL * 4);
  (void)ws_size; (void)n_in; (void)out_size;

  auto cvt = [&](const float* x, _Float16* y, int n) {
    k_cvt<<<(n + 1023) / 1024, 256, 0, stream>>>(x, y, n);
  };
  k_ropetab<<<(SEQ * 32) / 256, 256, 0, stream>>>(cs_tab, sn_tab);
  cvt(src, src_h, ntok * DMODEL);
  cvt(wq_a, wq_a_h, QRANK * DMODEL);
  cvt(wq_b, wq_b_h, NHEADS * QHD * QRANK);
  cvt(wkv_a, wkv_a_h, (KVRANK + ROPED) * DMODEL);
  cvt(wkv_b, wkv_b_h, NHEADS * 128 * KVRANK);
  cvt(wo, wo_h, DMODEL * DMODEL);
  k_absT<<<64, 256, 0, stream>>>(wkv_b, absT);

  k_ckv<<<ntok / 128, 256, 0, stream>>>(src_h, wkv_a_h, kv_ln_w, cs_tab, sn_tab,
                                        c_lat, c_latT, k_pe, ntok);
  k_qa<<<ntok / 128, 256, 0, stream>>>(src_h, wq_a_h, q_a_ln_w, q_a_n, ntok);
  k_qb<<<ntok / 16, 256, 0, stream>>>(q_a_n, wq_b_h, absT, cs_tab, sn_tab,
                                      q_lat, q_pe, ntok);
  k_attn<<<ntok / 16, 256, 0, stream>>>(q_lat, q_pe, c_lat, c_latT, k_pe,
                                        wkv_b_h, attnctx, ntok);
  k_wo<<<(ntok / 16) * 12 / 8, 256, 0, stream>>>(attnctx, wo_h, src, yres, ntok);
  k_outnorm<<<ntok / 8, 256, 0, stream>>>(yres, norm_w, out, ntok);
}
